// TorchSpexModel_12068858101840
// MI455X (gfx1250) — compile-verified, matrix operand order checked by reading, not run on hardware
//
#include <hip/hip_runtime.h>

// ---------------------------------------------------------------------------
// Species-compacted expert MLP (960 -> 128 -> 128 -> 1), BF16 WMMA, gfx1250.
// Pipeline: init -> count -> scan -> scatter -> weight-cvt -> zero -> main.
// ---------------------------------------------------------------------------

#define N_ATOMS   50000
#define N_FEAT    960
#define HIDDEN    128
#define N_SPECIES 4
#define N_STRUCT  1000

#define SILU_GAIN 1.6765f
#define TILE_M    64                   // atoms per block (4 WMMA M-tiles)
#define N_TILES_M 786                  // ceil(50000/64) + 4 species pads
#define PERM_CAP  (N_TILES_M * TILE_M) // 50304 padded permutation slots
#define XROW      968                  // padded LDS row (halfs): 484 dw % 64 = 36
#define HROW      136                  // padded LDS row: 68 dw % 64 = 4

typedef __attribute__((ext_vector_type(4)))  float  v4f;
typedef __attribute__((ext_vector_type(8)))  float  v8f;
typedef __attribute__((ext_vector_type(4)))  __bf16 v4bf;
typedef __attribute__((ext_vector_type(8)))  __bf16 v8bf;
typedef __attribute__((ext_vector_type(16))) __bf16 v16bf;

static __device__ inline v16bf cat16(v8bf lo, v8bf hi) {
    return __builtin_shufflevector(lo, hi, 0,1,2,3,4,5,6,7,8,9,10,11,12,13,14,15);
}

static __device__ inline float silu_gain(float x) {
    return (x / (1.0f + __expf(-x))) * SILU_GAIN;
}

// aux layout (ints): [0..3]=counts, [4..7]=cursors, [8..12]=bases
// ---------------------------------------------------------------------------
__global__ void k_init(int* __restrict__ perm, int* __restrict__ aux) {
    for (int i = blockIdx.x * blockDim.x + threadIdx.x; i < PERM_CAP;
         i += gridDim.x * blockDim.x) {
        perm[i] = -1;
        if (i < 16) aux[i] = 0;
    }
}

__global__ void k_count(const int* __restrict__ species, int* __restrict__ aux) {
    int i = blockIdx.x * blockDim.x + threadIdx.x;
    if (i < N_ATOMS) atomicAdd(&aux[species[i]], 1);
}

__global__ void k_scan(int* __restrict__ aux) {     // 1 thread
    int b = 0;
    aux[8] = 0;
    for (int s = 0; s < N_SPECIES; ++s) {
        b += ((aux[s] + TILE_M - 1) / TILE_M) * TILE_M;   // 64-aligned segments
        aux[9 + s] = b;
    }
}

__global__ void k_scatter(const int* __restrict__ species,
                          int* __restrict__ perm, int* __restrict__ aux) {
    int i = blockIdx.x * blockDim.x + threadIdx.x;
    if (i < N_ATOMS) {
        int s = species[i];
        int pos = aux[8 + s] + atomicAdd(&aux[4 + s], 1);
        perm[pos] = i;
    }
}

// ---------------------------------------------------------------------------
__global__ void k_prep_weights(const float* __restrict__ w1,
                               const float* __restrict__ w2,
                               __bf16* __restrict__ w1b,
                               __bf16* __restrict__ w2b) {
    const int n1 = N_SPECIES * HIDDEN * N_FEAT;   // 491520
    const int n2 = N_SPECIES * HIDDEN * HIDDEN;   // 65536
    for (int i = blockIdx.x * blockDim.x + threadIdx.x; i < n1 + n2;
         i += gridDim.x * blockDim.x) {
        if (i < n1) w1b[i] = (__bf16)w1[i];
        else        w2b[i - n1] = (__bf16)w2[i - n1];
    }
}

__global__ void k_zero_out(float* __restrict__ out, int n) {
    int i = blockIdx.x * blockDim.x + threadIdx.x;
    if (i < n) out[i] = 0.0f;
}

// ---------------------------------------------------------------------------
// Main: one block = 64 atoms of a single species; 8 wave32s, each owning a
// 16-wide hidden slice and 4 M-tiles (B-fragment reused by 4 WMMAs).
// ---------------------------------------------------------------------------
__global__ void __launch_bounds__(256)
k_expert_mlp(const float* __restrict__ features,
             const int*   __restrict__ structure_ids,
             const int*   __restrict__ perm,
             const int*   __restrict__ aux,
             const __bf16* __restrict__ w1b,
             const __bf16* __restrict__ w2b,
             const float* __restrict__ w3,
             const float* __restrict__ comp_coef,
             float* __restrict__ out) {
    __shared__ alignas(16) __bf16 Xs[TILE_M][XROW];   // bf16 feature tile
    __shared__ alignas(16) __bf16 Hs[TILE_M][HROW];   // layer-1 activations
    __shared__ alignas(16) __bf16 H2[TILE_M][HROW];   // layer-2 activations
    __shared__ int atom_of_row[TILE_M];

    const int tb = blockIdx.x * TILE_M;
    const int* bases = aux + 8;
    if (tb >= bases[N_SPECIES]) return;               // uniform early exit
    int sp = 0;
    while (sp < N_SPECIES - 1 && tb >= bases[sp + 1]) ++sp;

    const int tid   = threadIdx.x;
    const int wave  = tid >> 5;            // 0..7 -> hidden slice [16w,16w+16)
    const int lane  = tid & 31;
    const int row   = lane & 15;           // A-row-in-tile / B-col
    const int koff  = (lane < 16) ? 0 : 8; // 16-bit WMMA K striping
    const int mhi   = (lane < 16) ? 0 : 8; // C/D M striping
    const int hbase = wave * 16;

    // ---- stage row->atom map, then gather features as 16B chunks -----------
    if (tid < TILE_M) atom_of_row[tid] = perm[tb + tid];
    __syncthreads();

    if (tid < N_FEAT / 4) {                // 240 chunks of float4 per row
        const int c4 = tid * 4;
#pragma unroll 4
        for (int r = 0; r < TILE_M; ++r) {
            int atom = atom_of_row[r];
            v4bf b = {};
            if (atom >= 0) {
                v4f f = *(const v4f*)(features + (size_t)atom * N_FEAT + c4);
                b[0] = (__bf16)f[0]; b[1] = (__bf16)f[1];
                b[2] = (__bf16)f[2]; b[3] = (__bf16)f[3];
            }
            *(v4bf*)(&Xs[r][c4]) = b;
        }
    }
    __syncthreads();

    const float inv_s1 = 0.032274861f;     // 1/sqrt(960)
    const float inv_s2 = 0.088388348f;     // 1/sqrt(128)

    // ---------------- layer 1: [64x960] x [960x16] --------------------------
    v8f acc[4];
#pragma unroll
    for (int mt = 0; mt < 4; ++mt) acc[mt] = {};
    {
        const __bf16* w1row = w1b + ((size_t)(sp * HIDDEN + hbase + row)) * N_FEAT;
        for (int k0 = 0; k0 < N_FEAT; k0 += 32) {
            v16bf B = cat16(*(const v8bf*)(w1row + k0 + koff),
                            *(const v8bf*)(w1row + k0 + 16 + koff));
#pragma unroll
            for (int mt = 0; mt < 4; ++mt) {
                const __bf16* xr = &Xs[mt * 16 + row][0];
                v16bf A = cat16(*(const v8bf*)(xr + k0 + koff),
                                *(const v8bf*)(xr + k0 + 16 + koff));
                acc[mt] = __builtin_amdgcn_wmma_f32_16x16x32_bf16(
                    false, A, false, B, (short)0, acc[mt], false, false);
            }
        }
    }
#pragma unroll
    for (int mt = 0; mt < 4; ++mt)
#pragma unroll
        for (int r = 0; r < 8; ++r)
            Hs[mt * 16 + r + mhi][hbase + row] =
                (__bf16)silu_gain(acc[mt][r] * inv_s1);
    __syncthreads();

    // ---------------- layer 2: [64x128] x [128x16] --------------------------
    v8f acc2[4];
#pragma unroll
    for (int mt = 0; mt < 4; ++mt) acc2[mt] = {};
    {
        const __bf16* w2row = w2b + ((size_t)(sp * HIDDEN + hbase + row)) * HIDDEN;
#pragma unroll
        for (int k0 = 0; k0 < HIDDEN; k0 += 32) {
            v16bf B = cat16(*(const v8bf*)(w2row + k0 + koff),
                            *(const v8bf*)(w2row + k0 + 16 + koff));
#pragma unroll
            for (int mt = 0; mt < 4; ++mt) {
                const __bf16* hr = &Hs[mt * 16 + row][0];
                v16bf A = cat16(*(const v8bf*)(hr + k0 + koff),
                                *(const v8bf*)(hr + k0 + 16 + koff));
                acc2[mt] = __builtin_amdgcn_wmma_f32_16x16x32_bf16(
                    false, A, false, B, (short)0, acc2[mt], false, false);
            }
        }
    }
#pragma unroll
    for (int mt = 0; mt < 4; ++mt)
#pragma unroll
        for (int r = 0; r < 8; ++r)
            H2[mt * 16 + r + mhi][hbase + row] =
                (__bf16)silu_gain(acc2[mt][r] * inv_s2);
    __syncthreads();

    // ---------------- layer 3 + scatter (waves 0..3, 16 atoms each) ---------
    if (wave < 4) {
        const float* w3s = w3 + sp * HIDDEN;
        const int mrow = wave * 16 + (lane & 15);
        const int half = (lane >> 4) * 64;
        float sum = 0.0f;
#pragma unroll 8
        for (int j = 0; j < 64; ++j)
            sum += (float)H2[mrow][half + j] * w3s[half + j];
        sum += __shfl_xor(sum, 16, 32);    // combine the two K-halves
        if (lane < 16) {
            int atom = atom_of_row[mrow];
            if (atom >= 0)
                atomicAdd(&out[structure_ids[atom]],
                          sum * inv_s2 + comp_coef[sp]);
        }
    }
}

// ---------------------------------------------------------------------------
extern "C" void kernel_launch(void* const* d_in, const int* in_sizes, int n_in,
                              void* d_out, int out_size, void* d_ws, size_t ws_size,
                              hipStream_t stream) {
    const float* features      = (const float*)d_in[0];
    const int*   species_idx   = (const int*)d_in[1];
    const int*   structure_ids = (const int*)d_in[2];
    const float* W1            = (const float*)d_in[3];
    const float* W2            = (const float*)d_in[4];
    const float* W3            = (const float*)d_in[5];
    const float* comp_coef     = (const float*)d_in[6];
    float*       out           = (float*)d_out;

    // workspace layout
    char* ws = (char*)d_ws;
    __bf16* w1b  = (__bf16*)ws;                                   //   983,040 B
    __bf16* w2b  = (__bf16*)(ws + 983040);                        //   131,072 B
    int*    aux  = (int*)   (ws + 983040 + 131072);               //        64 B
    int*    perm = (int*)   (ws + 983040 + 131072 + 64);          //   201,216 B

    k_init   <<<196, 256, 0, stream>>>(perm, aux);
    k_count  <<<(N_ATOMS + 255) / 256, 256, 0, stream>>>(species_idx, aux);
    k_scan   <<<1, 1, 0, stream>>>(aux);
    k_scatter<<<(N_ATOMS + 255) / 256, 256, 0, stream>>>(species_idx, perm, aux);
    k_prep_weights<<<2176, 256, 0, stream>>>(W1, W2, w1b, w2b);
    k_zero_out<<<(N_STRUCT + 255) / 256, 256, 0, stream>>>(out, N_STRUCT);
    k_expert_mlp<<<N_TILES_M, 256, 0, stream>>>(
        features, structure_ids, perm, aux, w1b, w2b, W3, comp_coef, out);
}
